// MySeq2Seq_17033840296273
// MI455X (gfx1250) — compile-verified, matrix operand order checked by reading, not run on hardware
//
#include <hip/hip_runtime.h>
#include <hip/hip_bf16.h>

constexpr int BATCH = 64, SEQLEN = 256, DECL = 64, EMB = 256, HID = 512;
constexpr int G4 = 2048;       // 4*HID
constexpr int VOC = 214, VPAD = 224;
constexpr int SS2 = 512;       // 2*SEQLEN (attention memory length)
constexpr int DD2 = 1024;      // 2*HID
constexpr int KCAT = 768;      // EMB + HID combined GEMM K
constexpr int START_TOK = 213;
constexpr int KPAN = 128;      // K panel staged in LDS
constexpr int LDP  = 136;      // padded LDS row stride (halves): 272B = 68 dwords, 68%64=4

typedef _Float16 h16;
typedef __attribute__((ext_vector_type(16))) _Float16 v16h;
typedef __attribute__((ext_vector_type(8)))  _Float16 v8h;
typedef __attribute__((ext_vector_type(8)))  float    v8f;

__device__ __forceinline__ float sigm(float x) { return 1.0f / (1.0f + __expf(-x)); }

// 16x32 f16 fragment load (ISA 7.12.2 lane pattern): lane L -> row L%16,
// kbase=(L/16)*8; two contiguous 8-half (16B) runs at kbase and kbase+16.
__device__ __forceinline__ v16h load_frag(const h16* __restrict__ p, int ld) {
  int lane = threadIdx.x & 31;
  const h16* q = p + (lane & 15) * ld + ((lane >> 4) << 3);
  v8h lo = *(const v8h*)q;
  v8h hi = *(const v8h*)(q + 16);
  v16h o;
#pragma unroll
  for (int j = 0; j < 8; ++j) { o[j] = lo[j]; o[j + 8] = hi[j]; }
  return o;
}

__device__ __forceinline__ v8f wmma16(v16h a, v16h b, v8f c) {
  return __builtin_amdgcn_wmma_f32_16x16x32_f16(false, a, false, b, (short)0, c,
                                                false, false);
}

// CDNA5 async copy: 16B global -> LDS, tracked by ASYNCcnt (ISA 15.18.3 op 98).
__device__ __forceinline__ void async_b128(h16* lds, const h16* g) {
  unsigned loff = (unsigned)(size_t)lds;   // flat LDS addr: offset in low 32 bits
  asm volatile("global_load_async_to_lds_b128 %0, %1, off"
               :: "v"(loff), "v"(g) : "memory");
}
__device__ __forceinline__ void wait_async() {
  asm volatile("s_wait_asynccnt 0x0" ::: "memory");
}

// ---------------- setup kernels ----------------

__global__ void k_pack_w(const float* __restrict__ Wih, const float* __restrict__ Whh,
                         h16* __restrict__ dst) {
  int idx = blockIdx.x * blockDim.x + threadIdx.x;
  if (idx >= G4 * KCAT) return;
  int n = idx / KCAT, k = idx - n * KCAT;
  float v = (k < EMB) ? Wih[n * EMB + k] : Whh[n * HID + (k - EMB)];
  dst[idx] = (h16)v;
}

__global__ void k_pack_attnw(const float* __restrict__ W, h16* __restrict__ dst) {
  int idx = blockIdx.x * blockDim.x + threadIdx.x;
  if (idx >= VPAD * G4) return;
  int n = idx >> 11;
  dst[idx] = (n < VOC) ? (h16)W[idx] : (h16)0.0f;
}

__global__ void k_embed(const int* __restrict__ tok, const float* __restrict__ tab,
                        h16* __restrict__ dst, int n) {
  int idx = blockIdx.x * blockDim.x + threadIdx.x;
  if (idx >= n) return;
  int t = idx >> 8, e = idx & 255;
  dst[idx] = (h16)tab[tok[t] * EMB + e];
}

__global__ void k_embed_dec(const int* __restrict__ labels, const float* __restrict__ tab,
                            h16* __restrict__ dst) {
  int idx = blockIdx.x * blockDim.x + threadIdx.x;
  if (idx >= BATCH * DECL * EMB) return;
  int b = idx >> 14, t = (idx >> 8) & 63, e = idx & 255;
  int tok = (t == 0) ? START_TOK : labels[b * DECL + t - 1];
  dst[idx] = (h16)tab[tok * EMB + e];
}

// ---------------- multi-chain LSTM step ----------------

struct StepArgs {
  const h16*   emb[4];
  const h16*   hF16[4];
  const h16*   W[4];
  const float* bias[4];
  float*       g[4];
  float*       c[4];
  float*       h[4];
  h16*         hf[4];
  h16*         encH[4];     // null if chain does not write attention memory
  int rev[4];
  int segOff[4];
  int dimOff[4];
  int t, embStride, seqLen;
};

__global__ void k_zero4(StepArgs sa) {
  int chain = blockIdx.y;
  int idx = blockIdx.x * blockDim.x + threadIdx.x;
  if (idx >= BATCH * HID) return;
  sa.h[chain][idx] = 0.f; sa.c[chain][idx] = 0.f; sa.hf[chain][idx] = (h16)0.f;
}

// g = [x_t | h] @ [Wih|Whh]^T + bias for gridDim.y chains in parallel.
// M=64, N=2048, K=768. Block: 256 thr = 8 waves, each wave 2x2 tiles;
// block covers all M x 128 columns; A+B k-panels staged in LDS via async copy.
__global__ void k_lstm_gemm4(StepArgs sa) {
  __shared__ h16 shA[64 * LDP];
  __shared__ h16 shB[128 * LDP];
  int chain = blockIdx.y;
  int p = sa.rev[chain] ? (sa.seqLen - 1 - sa.t) : sa.t;
  const h16* emb = sa.emb[chain] + p * EMB;
  const h16* hF  = sa.hF16[chain];
  const h16* W   = sa.W[chain];
  int tid = threadIdx.x;
  int wave = tid >> 5, lane = tid & 31;
  int mg = wave & 1;            // 2 m-groups of 32 rows
  int ng = wave >> 1;           // 4 n-groups of 32 cols
  int nColBase = blockIdx.x * 128;
  v8f acc[4] = {{}, {}, {}, {}};
  for (int kp = 0; kp < KCAT; kp += KPAN) {
    __syncthreads();            // previous panel fully consumed
    const h16* srcA; int lda;
    if (kp < EMB) { srcA = emb + kp; lda = sa.embStride; }
    else          { srcA = hF + (kp - EMB); lda = HID; }
    for (int i = tid; i < 64 * (KPAN / 8); i += 256) {      // 16B chunks of A
      int row = i >> 4, kc = (i & 15) << 3;
      async_b128(&shA[row * LDP + kc], srcA + row * lda + kc);
    }
    for (int i = tid; i < 128 * (KPAN / 8); i += 256) {     // 16B chunks of B
      int col = i >> 4, kc = (i & 15) << 3;
      const h16* src = W + (size_t)(nColBase + col) * KCAT + kp + kc;
      __builtin_prefetch(src + KPAN, 0, 1);
      async_b128(&shB[col * LDP + kc], src);
    }
    wait_async();               // this wave's async copies done
    __syncthreads();            // all waves' copies visible
#pragma unroll
    for (int kk = 0; kk < KPAN; kk += 32) {
      v16h a0 = load_frag(&shA[(mg * 32 + 0)  * LDP + kk], LDP);
      v16h a1 = load_frag(&shA[(mg * 32 + 16) * LDP + kk], LDP);
      v16h b0 = load_frag(&shB[(ng * 32 + 0)  * LDP + kk], LDP);
      v16h b1 = load_frag(&shB[(ng * 32 + 16) * LDP + kk], LDP);
      acc[0] = wmma16(a0, b0, acc[0]);
      acc[1] = wmma16(a0, b1, acc[1]);
      acc[2] = wmma16(a1, b0, acc[2]);
      acc[3] = wmma16(a1, b1, acc[3]);
    }
  }
  float* g = sa.g[chain];
  const float* bias = sa.bias[chain];
  int colBase = nColBase + ng * 32 + (lane & 15);
  int rowBase = mg * 32 + ((lane >> 4) << 3);
#pragma unroll
  for (int mi = 0; mi < 2; ++mi)
#pragma unroll
    for (int ni = 0; ni < 2; ++ni) {
      int col = colBase + ni * 16;
      int rb  = rowBase + mi * 16;
      float bv = bias[col];
#pragma unroll
      for (int r = 0; r < 8; ++r) g[(rb + r) * G4 + col] = acc[mi * 2 + ni][r] + bv;
    }
}

__global__ void k_lstm_gate4(StepArgs sa) {
  int chain = blockIdx.y;
  int idx = blockIdx.x * blockDim.x + threadIdx.x;
  if (idx >= BATCH * HID) return;
  int b = idx >> 9, n = idx & 511;
  const float* gb = sa.g[chain] + b * G4;
  float gi = sigm(gb[n]);
  float gf = sigm(gb[HID + n]);
  float gg = tanhf(gb[2 * HID + n]);
  float go = sigm(gb[3 * HID + n]);
  float cn = gf * sa.c[chain][idx] + gi * gg;
  float hn = go * tanhf(cn);
  sa.c[chain][idx] = cn; sa.h[chain][idx] = hn; sa.hf[chain][idx] = (h16)hn;
  h16* eH = sa.encH[chain];
  if (eH) {
    int p = sa.rev[chain] ? (sa.seqLen - 1 - sa.t) : sa.t;
    eH[((size_t)b * SS2 + sa.segOff[chain] + p) * DD2 + sa.dimOff[chain] + n] = (h16)hn;
  }
}

// hidden/cell FC: dst[d,b,off+j] = relu(hin_d[b]) @ W^T + bias, j<HID/2
__global__ void k_state_fc(const float* __restrict__ hInF, const float* __restrict__ hInB,
                           const float* __restrict__ W, const float* __restrict__ bias,
                           float* __restrict__ dst, h16* __restrict__ dstF16, int off) {
  int idx = blockIdx.x * blockDim.x + threadIdx.x;
  if (idx >= 2 * BATCH * 256) return;
  int d = idx >> 14, b = (idx >> 8) & 63, j = idx & 255;
  const float* hin = (d == 0 ? hInF : hInB) + b * HID;
  float acc = bias[j];
  for (int k = 0; k < HID; ++k) acc += fmaxf(hin[k], 0.f) * W[j * HID + k];
  int o = (d * BATCH + b) * HID + off + j;
  dst[o] = acc;
  if (dstF16) dstF16[o] = (h16)acc;
}

// ---------------- attention ----------------

__global__ void k_attn_scores(const h16* __restrict__ encH, const float* __restrict__ decH,
                              float* __restrict__ scores) {
  int b = blockIdx.x;
  int wave = threadIdx.x >> 5, lane = threadIdx.x & 31;
  const float* hf = decH + b * HID;
  const float* hb = decH + (BATCH + b) * HID;
  for (int s = wave; s < SS2; s += 8) {
    const h16* e = encH + ((size_t)b * SS2 + s) * DD2;
    float acc = 0.f;
    for (int k = lane; k < HID; k += 32)
      acc += (float)e[k] * hf[k] + (float)e[HID + k] * hb[k];
#pragma unroll
    for (int off = 16; off; off >>= 1) acc += __shfl_xor(acc, off, 32);
    if (lane == 0) scores[b * SS2 + s] = acc;
  }
}

__global__ void k_attn_av_y(const float* __restrict__ scores, const h16* __restrict__ encH,
                            const float* __restrict__ decH, h16* __restrict__ y) {
  __shared__ float sm[256];
  __shared__ float w[SS2];
  int b = blockIdx.x, tid = threadIdx.x;
  float m = -1e30f;
  for (int s = tid; s < SS2; s += 256) m = fmaxf(m, scores[b * SS2 + s]);
  sm[tid] = m; __syncthreads();
  for (int st = 128; st; st >>= 1) { if (tid < st) sm[tid] = fmaxf(sm[tid], sm[tid + st]); __syncthreads(); }
  m = sm[0]; __syncthreads();
  float sum = 0.f;
  for (int s = tid; s < SS2; s += 256) { float e = __expf(scores[b * SS2 + s] - m); w[s] = e; sum += e; }
  sm[tid] = sum; __syncthreads();
  for (int st = 128; st; st >>= 1) { if (tid < st) sm[tid] += sm[tid + st]; __syncthreads(); }
  float inv = 1.f / sm[0];
  __syncthreads();
#pragma unroll
  for (int d0 = 0; d0 < 4; ++d0) {
    int d = tid + d0 * 256;
    float acc = 0.f;
    for (int s = 0; s < SS2; ++s) acc += w[s] * (float)encH[((size_t)b * SS2 + s) * DD2 + d];
    acc *= inv;
    y[b * G4 + DD2 + d] = (h16)fmaxf(tanhf(acc), 0.f);
  }
  for (int d = tid; d < DD2; d += 256) {
    float hv = (d < HID) ? decH[b * HID + d] : decH[(BATCH + b) * HID + (d - HID)];
    y[b * G4 + d] = (h16)fmaxf(tanhf(hv), 0.f);
  }
}

// logits = y [64,2048] @ attnW^T [2048, VPAD] + bias. 4x14 = 56 tiles, 7 blocks.
__global__ void k_attn_gemm(const h16* __restrict__ y, const h16* __restrict__ Wp,
                            const float* __restrict__ bias, float* __restrict__ logits) {
  int wave = (blockIdx.x * blockDim.x + threadIdx.x) >> 5;
  int mt = wave & 3, nt = wave >> 2;   // nt < 14
  int lane = threadIdx.x & 31;
  v8f acc = {};
  const h16* wrow = Wp + nt * 16 * G4;
  for (int k0 = 0; k0 < G4; k0 += 32) {
    __builtin_prefetch(wrow + k0 + 64, 0, 1);
    v16h a  = load_frag(y + mt * 16 * G4 + k0, G4);
    v16h bf = load_frag(wrow + k0, G4);
    acc = wmma16(a, bf, acc);
  }
  int col = nt * 16 + (lane & 15);
  int rbase = mt * 16 + ((lane >> 4) << 3);
  float bv = (col < VOC) ? bias[col] : 0.f;
#pragma unroll
  for (int r = 0; r < 8; ++r) logits[(rbase + r) * VPAD + col] = acc[r] + bv;
}

__global__ void k_logsoftmax(const float* __restrict__ logits, float* __restrict__ out) {
  __shared__ float sm[256];
  int b = blockIdx.x, tid = threadIdx.x;
  float v = (tid < VOC) ? logits[b * VPAD + tid] : -1e30f;
  sm[tid] = v; __syncthreads();
  for (int st = 128; st; st >>= 1) { if (tid < st) sm[tid] = fmaxf(sm[tid], sm[tid + st]); __syncthreads(); }
  float m = sm[0]; __syncthreads();
  float e = (tid < VOC) ? __expf(v - m) : 0.f;
  sm[tid] = e; __syncthreads();
  for (int st = 128; st; st >>= 1) { if (tid < st) sm[tid] += sm[tid + st]; __syncthreads(); }
  float lse = m + __logf(sm[0]);
  if (tid < VOC) out[b * VOC + tid] = v - lse;
}

// ---------------- host orchestration ----------------

extern "C" void kernel_launch(void* const* d_in, const int* in_sizes, int n_in,
                              void* d_out, int out_size, void* d_ws, size_t ws_size,
                              hipStream_t stream) {
  const int* prefix  = (const int*)d_in[0];
  const int* postfix = (const int*)d_in[1];
  const int* labels  = (const int*)d_in[2];
  const float* pre_emb   = (const float*)d_in[4];
  const float* pre_Wih_f = (const float*)d_in[5];
  const float* pre_Whh_f = (const float*)d_in[6];
  const float* pre_b_f   = (const float*)d_in[7];
  const float* pre_Wih_b = (const float*)d_in[8];
  const float* pre_Whh_b = (const float*)d_in[9];
  const float* pre_b_b   = (const float*)d_in[10];
  const float* pre_hfc_W = (const float*)d_in[11];
  const float* pre_hfc_b = (const float*)d_in[12];
  const float* pre_cfc_W = (const float*)d_in[13];
  const float* pre_cfc_b = (const float*)d_in[14];
  const float* post_emb   = (const float*)d_in[15];
  const float* post_Wih_f = (const float*)d_in[16];
  const float* post_Whh_f = (const float*)d_in[17];
  const float* post_b_f   = (const float*)d_in[18];
  const float* post_Wih_b = (const float*)d_in[19];
  const float* post_Whh_b = (const float*)d_in[20];
  const float* post_b_b   = (const float*)d_in[21];
  const float* post_hfc_W = (const float*)d_in[22];
  const float* post_hfc_b = (const float*)d_in[23];
  const float* post_cfc_W = (const float*)d_in[24];
  const float* post_cfc_b = (const float*)d_in[25];
  const float* dec_emb   = (const float*)d_in[26];
  const float* dec_Wih_f = (const float*)d_in[27];
  const float* dec_Whh_f = (const float*)d_in[28];
  const float* dec_b_f   = (const float*)d_in[29];
  const float* dec_Wih_b = (const float*)d_in[30];
  const float* dec_Whh_b = (const float*)d_in[31];
  const float* dec_b_b   = (const float*)d_in[32];
  const float* attn_W = (const float*)d_in[33];
  const float* attn_b = (const float*)d_in[34];
  float* out = (float*)d_out;

  char* wp = (char*)d_ws;
  auto alloc = [&](size_t bytes) -> void* {
    void* p = (void*)wp;
    wp += (bytes + 255) & ~(size_t)255;
    return p;
  };
  h16* Wc[6];
  for (int i = 0; i < 6; ++i) Wc[i] = (h16*)alloc((size_t)G4 * KCAT * 2);
  h16*   attnWp  = (h16*)alloc((size_t)VPAD * G4 * 2);
  h16*   embPre  = (h16*)alloc((size_t)BATCH * SEQLEN * EMB * 2);
  h16*   embPost = (h16*)alloc((size_t)BATCH * SEQLEN * EMB * 2);
  h16*   embDec  = (h16*)alloc((size_t)BATCH * DECL * EMB * 2);
  h16*   encH    = (h16*)alloc((size_t)BATCH * SS2 * DD2 * 2);
  float* g4[4]; float* h4[4]; float* c4[4]; h16* hf4[4];
  for (int i = 0; i < 4; ++i) {
    g4[i]  = (float*)alloc((size_t)BATCH * G4 * 4);
    h4[i]  = (float*)alloc((size_t)BATCH * HID * 4);
    c4[i]  = (float*)alloc((size_t)BATCH * HID * 4);
    hf4[i] = (h16*)alloc((size_t)BATCH * HID * 2);
  }
  float* decH    = (float*)alloc((size_t)2 * BATCH * HID * 4);
  float* decC    = (float*)alloc((size_t)2 * BATCH * HID * 4);
  h16*   decHF16 = (h16*)alloc((size_t)2 * BATCH * HID * 2);
  float* scores  = (float*)alloc((size_t)BATCH * SS2 * 4);
  h16*   yBuf    = (h16*)alloc((size_t)BATCH * G4 * 2);
  float* logits  = (float*)alloc((size_t)BATCH * VPAD * 4);

  dim3 t256(256);
  k_pack_w<<<(G4 * KCAT + 255) / 256, t256, 0, stream>>>(pre_Wih_f, pre_Whh_f, Wc[0]);
  k_pack_w<<<(G4 * KCAT + 255) / 256, t256, 0, stream>>>(pre_Wih_b, pre_Whh_b, Wc[1]);
  k_pack_w<<<(G4 * KCAT + 255) / 256, t256, 0, stream>>>(post_Wih_f, post_Whh_f, Wc[2]);
  k_pack_w<<<(G4 * KCAT + 255) / 256, t256, 0, stream>>>(post_Wih_b, post_Whh_b, Wc[3]);
  k_pack_w<<<(G4 * KCAT + 255) / 256, t256, 0, stream>>>(dec_Wih_f, dec_Whh_f, Wc[4]);
  k_pack_w<<<(G4 * KCAT + 255) / 256, t256, 0, stream>>>(dec_Wih_b, dec_Whh_b, Wc[5]);
  k_pack_attnw<<<(VPAD * G4 + 255) / 256, t256, 0, stream>>>(attn_W, attnWp);
  int nPre = BATCH * SEQLEN * EMB;
  k_embed<<<(nPre + 255) / 256, t256, 0, stream>>>(prefix, pre_emb, embPre, nPre);
  k_embed<<<(nPre + 255) / 256, t256, 0, stream>>>(postfix, post_emb, embPost, nPre);
  k_embed_dec<<<(BATCH * DECL * EMB + 255) / 256, t256, 0, stream>>>(labels, dec_emb, embDec);

  // --- encoders: all 4 chains (pre_f, pre_b, post_f, post_b) in parallel ---
  StepArgs enc{};
  {
    const h16*   embs[4]   = {embPre, embPre, embPost, embPost};
    const h16*   Ws[4]     = {Wc[0], Wc[1], Wc[2], Wc[3]};
    const float* biases[4] = {pre_b_f, pre_b_b, post_b_f, post_b_b};
    int revs[4] = {0, 1, 0, 1};
    int segs[4] = {0, 0, SEQLEN, SEQLEN};
    int dims[4] = {0, HID, 0, HID};
    for (int ch = 0; ch < 4; ++ch) {
      enc.emb[ch] = embs[ch]; enc.hF16[ch] = hf4[ch]; enc.W[ch] = Ws[ch];
      enc.bias[ch] = biases[ch]; enc.g[ch] = g4[ch];
      enc.c[ch] = c4[ch]; enc.h[ch] = h4[ch]; enc.hf[ch] = hf4[ch];
      enc.encH[ch] = encH; enc.rev[ch] = revs[ch];
      enc.segOff[ch] = segs[ch]; enc.dimOff[ch] = dims[ch];
    }
    enc.embStride = SEQLEN * EMB; enc.seqLen = SEQLEN;
  }
  k_zero4<<<dim3(128, 4), t256, 0, stream>>>(enc);
  for (int t = 0; t < SEQLEN; ++t) {
    enc.t = t;
    k_lstm_gemm4<<<dim3(16, 4), t256, 0, stream>>>(enc);
    k_lstm_gate4<<<dim3(128, 4), t256, 0, stream>>>(enc);
  }
  k_state_fc<<<128, t256, 0, stream>>>(h4[0], h4[1], pre_hfc_W, pre_hfc_b, decH, decHF16, 0);
  k_state_fc<<<128, t256, 0, stream>>>(c4[0], c4[1], pre_cfc_W, pre_cfc_b, decC, nullptr, 0);
  k_state_fc<<<128, t256, 0, stream>>>(h4[2], h4[3], post_hfc_W, post_hfc_b, decH, decHF16, 256);
  k_state_fc<<<128, t256, 0, stream>>>(c4[2], c4[3], post_cfc_W, post_cfc_b, decC, nullptr, 256);

  // --- decoder: 2 cells in parallel + attention ---
  StepArgs dec{};
  {
    const float* biases[2] = {dec_b_f, dec_b_b};
    for (int ch = 0; ch < 2; ++ch) {
      dec.emb[ch] = embDec; dec.hF16[ch] = decHF16 + ch * BATCH * HID;
      dec.W[ch] = Wc[4 + ch]; dec.bias[ch] = biases[ch]; dec.g[ch] = g4[ch];
      dec.c[ch] = decC + ch * BATCH * HID; dec.h[ch] = decH + ch * BATCH * HID;
      dec.hf[ch] = decHF16 + ch * BATCH * HID;
      dec.encH[ch] = nullptr; dec.rev[ch] = 0; dec.segOff[ch] = 0; dec.dimOff[ch] = 0;
    }
    dec.embStride = DECL * EMB; dec.seqLen = DECL;
  }
  for (int t = 0; t < DECL; ++t) {
    dec.t = t;
    k_lstm_gemm4<<<dim3(16, 2), t256, 0, stream>>>(dec);   // reads old h
    k_lstm_gate4<<<dim3(128, 2), t256, 0, stream>>>(dec);  // writes new h,c
    k_attn_scores<<<BATCH, t256, 0, stream>>>(encH, decH, scores);
    k_attn_av_y<<<BATCH, t256, 0, stream>>>(scores, encH, decH, yBuf);
    k_attn_gemm<<<7, t256, 0, stream>>>(yBuf, attnWp, attn_b, logits);
    k_logsoftmax<<<BATCH, t256, 0, stream>>>(logits, out + (size_t)t * BATCH * VOC);
  }
}
